// InstanceDecoder_59854664237372
// MI455X (gfx1250) — matrix operand order repaired and checked by reading, compiled, not run on hardware
//
#include <hip/hip_runtime.h>

// ---------------- problem constants ----------------
#define N_HITS   16384
#define FEAT     128
#define CDIM     8
#define JSPLIT   16
#define BETA_TH  0.1f

typedef __attribute__((ext_vector_type(16))) _Float16 v16h;
typedef __attribute__((ext_vector_type(8)))  float    v8f;

// ---------------- workspace layout (bytes) ----------------
// astage: per row, 32 halves in contiguous K order (K = 16*h + s):
//   [0..7]  = -2*ox (K=0..7)   [8] = sq (K=8)   [9] = 1.0 (K=9)
//   [10..31] = 0
// bstage: per column, 16 halves: [ox(8), 1.0, sq_or_big, 0 x6]
#define OFF_A   ((size_t)0)                                  // N*32*2 = 1,048,576 B
#define OFF_B   ((size_t)(N_HITS * 32 * 2))                  // N*16*2 =   524,288 B
#define OFF_Z   (OFF_B + (size_t)(N_HITS * 16 * 2))          // 32 B zero block
#define OFF_PV  (OFF_Z + (size_t)1024)                       // JSPLIT*N floats
#define OFF_PI  (OFF_PV + (size_t)JSPLIT * N_HITS * 4)       // JSPLIT*N ints

// ================= kernel 1: projections + operand staging =================
__global__ void __launch_bounds__(256)
k1_project(const float* __restrict__ x,
           const float* __restrict__ beta_w,
           const float* __restrict__ beta_b,
           const float* __restrict__ coord_w,
           const float* __restrict__ coord_b,
           float*    __restrict__ out,     // [N, 9] = (of, ox)
           _Float16* __restrict__ astage,  // [N, 32]
           _Float16* __restrict__ bstage,  // [N, 16]
           _Float16* __restrict__ zeroblk) // [16]
{
    const int lane = threadIdx.x & 31;
    const int wv   = threadIdx.x >> 5;
    const int row  = blockIdx.x * (blockDim.x >> 5) + wv;

    if (blockIdx.x == 0 && threadIdx.x < 16)
        zeroblk[threadIdx.x] = (_Float16)0.0f;

    float acc[1 + CDIM];
#pragma unroll
    for (int c = 0; c < 1 + CDIM; ++c) acc[c] = 0.0f;

    const float* xr = x + (size_t)row * FEAT;
#pragma unroll
    for (int q = 0; q < FEAT / 32; ++q) {
        const int k  = lane + 32 * q;
        const float xv = xr[k];
        acc[0] = fmaf(xv, beta_w[k], acc[0]);
#pragma unroll
        for (int c = 0; c < CDIM; ++c)
            acc[1 + c] = fmaf(xv, coord_w[c * FEAT + k], acc[1 + c]);
    }
#pragma unroll
    for (int off = 16; off > 0; off >>= 1) {
#pragma unroll
        for (int c = 0; c < 1 + CDIM; ++c)
            acc[c] += __shfl_xor(acc[c], off, 32);
    }

    if (lane == 0) {
        const float of = 1.0f / (1.0f + __expf(-(acc[0] + beta_b[0])));
        float oxc[CDIM];
        float sq = 0.0f;
#pragma unroll
        for (int c = 0; c < CDIM; ++c) {
            oxc[c] = acc[1 + c] + coord_b[c];
            sq = fmaf(oxc[c], oxc[c], sq);
        }
        float* orow = out + (size_t)row * (1 + CDIM);
        orow[0] = of;
#pragma unroll
        for (int c = 0; c < CDIM; ++c) orow[1 + c] = oxc[c];

        _Float16* ar = astage + (size_t)row * 32;
#pragma unroll
        for (int c = 0; c < CDIM; ++c) ar[c] = (_Float16)(-2.0f * oxc[c]);
#pragma unroll
        for (int c = CDIM; c < 16; ++c) ar[c] = (_Float16)0.0f;
        ar[8] = (_Float16)sq;
        ar[9] = (_Float16)1.0f;
#pragma unroll
        for (int c = 16; c < 32; ++c) ar[c] = (_Float16)0.0f;

        _Float16* br = bstage + (size_t)row * 16;
#pragma unroll
        for (int c = 0; c < CDIM; ++c) br[c] = (_Float16)oxc[c];
        br[8] = (_Float16)1.0f;
        br[9] = (of > BETA_TH) ? (_Float16)sq : (_Float16)30000.0f;  // mask fold
#pragma unroll
        for (int c = 10; c < 16; ++c) br[c] = (_Float16)0.0f;
    }
}

// ================= kernel 2: WMMA pairwise d^2 + running argmin =================
// One wave owns a 16-row strip and one of JSPLIT column ranges.
// d^2 comes straight out of V_WMMA_F32_16X16X32_F16 via the distance embedding:
//   A row = [-2*ox_i, sq_i, 1, 0...], B col = [ox_j, 1, sq_j|3e4, 0...]
// Software-pipelined 2 deep: load B[t+1]; wmma C[t]; min-scan C[t-1]
// (kills the loadcnt-0 stall and the WMMA->VALU hazard nops).

__device__ __forceinline__ void tile_minscan(const v8f c, const int jc,
                                             float (&minval)[8], int (&minidx)[8])
{
#pragma unroll
    for (int v = 0; v < 8; ++v) {
        const float d2 = c[v];               // no clamp: values only compared
        const bool upd = d2 < minval[v];     // strict < keeps lowest j on ties
        minval[v] = upd ? d2 : minval[v];
        minidx[v] = upd ? jc : minidx[v];
    }
}

__global__ void __launch_bounds__(256)
k2_pairmin(const _Float16* __restrict__ astage,
           const _Float16* __restrict__ bstage,
           const _Float16* __restrict__ zeroblk,
           float* __restrict__ pval,   // [JSPLIT, N]
           int*   __restrict__ pidx)   // [JSPLIT, N]
{
    const int  lane  = threadIdx.x & 31;
    const int  wv    = threadIdx.x >> 5;
    const int  m     = lane & 15;
    const bool hi    = lane >= 16;
    const int  ibase = (blockIdx.x * (blockDim.x >> 5) + wv) * 16;
    const int  split = blockIdx.y;
    const int  j0    = split * (N_HITS / JSPLIT);
    const int  TILES = N_HITS / JSPLIT / 16;   // 64

    // A operand (loop invariant): 16 halves per lane, contiguous K = 16*h + s
    const v16h A = *(const v16h*)(astage + (size_t)(ibase + m) * 32 + (hi ? 16 : 0));
    const v8f  zero = {};

    // per-lane streaming pointer: hi lanes stay parked on the 32B zero block
    // (they hold K=16..31, all zero); low lanes advance 512B per tile.
    const _Float16* bp      = hi ? zeroblk : (bstage + (size_t)(j0 + m) * 16);
    const size_t    bstride = hi ? 0 : (size_t)16 * 16;   // halves per j-tile

    float minval[8];
    int   minidx[8];
#pragma unroll
    for (int v = 0; v < 8; ++v) { minval[v] = 1.0f; minidx[v] = -1; }  // 1.0 = DIST_THRESH

    // ---- pipeline prologue: tiles 0 and 1 loaded, tile 0 mma issued ----
    v16h Bcur = *(const v16h*)bp;  bp += bstride;
    v16h Bnxt = *(const v16h*)bp;  bp += bstride;
    v8f  cpend = __builtin_amdgcn_wmma_f32_16x16x32_f16(
        false, A, false, Bcur, (short)0, zero, false, false);
    int  jpend = j0 + m;

    // ---- steady state: t = 1 .. TILES-2 ----
    for (int t = 1; t < TILES - 1; ++t) {
        __builtin_prefetch(bp + 2 * 256, 0, 1);          // 2 tiles ahead of B stream
        const v16h Bnn = *(const v16h*)bp;  bp += bstride;
        const v8f  cn  = __builtin_amdgcn_wmma_f32_16x16x32_f16(
            false, A, false, Bnxt, (short)0, zero, false, false);
        tile_minscan(cpend, jpend, minval, minidx);      // tile t-1 (no hazard wait)
        jpend += 16;
        cpend = cn;
        Bnxt  = Bnn;
    }

    // ---- epilogue: last mma + last two scans ----
    const v8f clast = __builtin_amdgcn_wmma_f32_16x16x32_f16(
        false, A, false, Bnxt, (short)0, zero, false, false);
    tile_minscan(cpend, jpend, minval, minidx);
    tile_minscan(clast, jpend + 16, minval, minidx);

    // reduce across the 16 lanes of each half (C layout: vgpr v = row v / v+8, lane%16 = col)
#pragma unroll
    for (int off = 1; off < 16; off <<= 1) {
#pragma unroll
        for (int v = 0; v < 8; ++v) {
            const float ov = __shfl_xor(minval[v], off, 32);
            const int   oi = __shfl_xor(minidx[v], off, 32);
            const bool take = (ov < minval[v]) ||
                              ((ov == minval[v]) && ((unsigned)oi < (unsigned)minidx[v]));
            minval[v] = take ? ov : minval[v];
            minidx[v] = take ? oi : minidx[v];
        }
    }

    if (m == 0) {
#pragma unroll
        for (int v = 0; v < 8; ++v) {
            const int row = ibase + v + (hi ? 8 : 0);
            pval[(size_t)split * N_HITS + row] = minval[v];
            pidx[(size_t)split * N_HITS + row] = minidx[v];
        }
    }
}

// ================= kernel 3: combine the JSPLIT partials =================
__global__ void __launch_bounds__(256)
k3_combine(const float* __restrict__ pval,
           const int*   __restrict__ pidx,
           float* __restrict__ inst_out)   // [N] instances, as float
{
    const int row = blockIdx.x * blockDim.x + threadIdx.x;
    if (row >= N_HITS) return;
    float    best = 1.0f;
    unsigned bi   = 0xFFFFFFFFu;            // -1
#pragma unroll
    for (int s = 0; s < JSPLIT; ++s) {
        const float    v = pval[(size_t)s * N_HITS + row];
        const unsigned i = (unsigned)pidx[(size_t)s * N_HITS + row];
        const bool take = (v < best) || ((v == best) && (i < bi));
        best = take ? v : best;
        bi   = take ? i : bi;
    }
    inst_out[row] = (float)(int)bi;
}

// ================= launcher =================
extern "C" void kernel_launch(void* const* d_in, const int* in_sizes, int n_in,
                              void* d_out, int out_size, void* d_ws, size_t ws_size,
                              hipStream_t stream)
{
    const float* x       = (const float*)d_in[0];
    const float* beta_w  = (const float*)d_in[1];
    const float* beta_b  = (const float*)d_in[2];
    const float* coord_w = (const float*)d_in[3];
    const float* coord_b = (const float*)d_in[4];

    float* out = (float*)d_out;                       // [N*9] then [N] instances
    char*  ws  = (char*)d_ws;

    _Float16* astage  = (_Float16*)(ws + OFF_A);
    _Float16* bstage  = (_Float16*)(ws + OFF_B);
    _Float16* zeroblk = (_Float16*)(ws + OFF_Z);
    float*    pval    = (float*)(ws + OFF_PV);
    int*      pidx    = (int*)(ws + OFF_PI);

    // kernel 1: 1 wave per row, 8 waves per block
    k1_project<<<N_HITS / 8, 256, 0, stream>>>(x, beta_w, beta_b, coord_w, coord_b,
                                               out, astage, bstage, zeroblk);

    // kernel 2: 8 waves/block * 16 rows/wave; y = column split (16384 waves total)
    dim3 g2(N_HITS / 16 / 8, JSPLIT);
    k2_pairmin<<<g2, 256, 0, stream>>>(astage, bstage, zeroblk, pval, pidx);

    // kernel 3: combine partials -> instances
    k3_combine<<<(N_HITS + 255) / 256, 256, 0, stream>>>(pval, pidx, out + (size_t)N_HITS * 9);
}
